// PolicyNetwork_76476187672805
// MI455X (gfx1250) — compile-verified, matrix operand order checked by reading, not run on hardware
//
#include <hip/hip_runtime.h>
#include <hip/hip_bf16.h>
#include <math.h>

#define DIM_EMBED 256
#define NUM_DAGS  1024
#define OPS_PER_DAG 100
#define TOTAL_OPS (NUM_DAGS * OPS_PER_DAG)   // 102400
#define NUM_WORKERS 64
#define H1V 32
#define H2V 16

typedef __attribute__((ext_vector_type(16))) __bf16 v16bf;
typedef __attribute__((ext_vector_type(8)))  float  v8f;

// ---------------- workspace layout (bytes) ----------------
#define WS_WT     0         // 32*256 bf16 (op_W1[0:256]^T)        = 16384
#define WS_C1     16384     // 32 f32  (z@W1_hi + b1, op branch)
#define WS_CP1    16512     // 32 f32  (z@prW1_hi + b1, pr branch)
#define WS_GMAX   16640     // f32
#define WS_GINV   16644     // f32
#define WS_PMAX   16896     // 256 f32
#define WS_PSUM   17920     // 256 f32
#define WS_YDOT1  18944     // 1024*32 f32 = 131072 (op branch)
#define WS_YDOTP  150016    // 1024*32 f32 = 131072 (pr branch)
// total = 281088 bytes

// ---------------- pass 1: fold z into biases, bf16-transpose W1_x ----------
__global__ void k_const(const float* __restrict__ z,
                        const float* __restrict__ opW1, const float* __restrict__ opB1,
                        const float* __restrict__ prW1, const float* __restrict__ prB1,
                        void* ws) {
  float* c1  = (float*)((char*)ws + WS_C1);
  float* cp1 = (float*)((char*)ws + WS_CP1);
  __bf16* wt = (__bf16*)((char*)ws + WS_WT);
  int t = threadIdx.x;
  if (t < H1V) {                        // op branch: rows [512,768) of op_W1 act on z
    float a = opB1[t];
    for (int e = 0; e < DIM_EMBED; ++e) a += z[e] * opW1[(2 * DIM_EMBED + e) * H1V + t];
    c1[t] = a;
  } else if (t < 2 * H1V) {             // pr branch: rows [257,513) of pr_W1 act on z
    int h = t - H1V;
    float a = prB1[h];
    for (int e = 0; e < DIM_EMBED; ++e) a += z[e] * prW1[(1 + DIM_EMBED + e) * H1V + h];
    cp1[h] = a;
  }
  // Wt[h][e] = bf16(op_W1[e][h]) : row-major W^T == column-major B (ISA B layout)
  for (int idx = t; idx < H1V * DIM_EMBED; idx += blockDim.x) {
    int h = idx >> 8, e = idx & 255;
    wt[h * DIM_EMBED + e] = (__bf16)opW1[e * H1V + h];
  }
}

// ---------------- pass 2: fold y into per-dag vectors, both branches -------
__global__ void k_ydot(const float* __restrict__ y,
                       const float* __restrict__ opW1,
                       const float* __restrict__ prW1,
                       void* ws) {
  __shared__ float sy[DIM_EMBED];
  int d = blockIdx.x, t = threadIdx.x;
  for (int i = t; i < DIM_EMBED; i += 64) sy[i] = y[d * DIM_EMBED + i];
  __syncthreads();
  if (t < H1V) {                        // op branch: rows [256,512) of op_W1
    float a = 0.f;
    for (int e = 0; e < DIM_EMBED; ++e) a += sy[e] * opW1[(DIM_EMBED + e) * H1V + t];
    ((float*)((char*)ws + WS_YDOT1))[d * H1V + t] = a;
  } else {                              // pr branch: rows [1,257) of pr_W1
    int h = t - H1V;
    float a = 0.f;
    for (int e = 0; e < DIM_EMBED; ++e) a += sy[e] * prW1[(1 + e) * H1V + h];
    ((float*)((char*)ws + WS_YDOTP))[d * H1V + h] = a;
  }
}

// ---------------- pass 3: WMMA GEMM x@W1_lo + MLP tail → logits -----------
// 800 blocks x 256 threads; wave w owns rows [blk*128 + w*16, +16)
__global__ void __launch_bounds__(256)
k_gemm(const float* __restrict__ x, const float* __restrict__ op_msk,
       const float* __restrict__ opW2, const float* __restrict__ opB2,
       const float* __restrict__ opW3, const float* __restrict__ opB3,
       const void* __restrict__ ws, float* __restrict__ out_logits) {
  __shared__ float h1lds[128][33];      // padded vs bank conflicts

  const __bf16* wt   = (const __bf16*)((const char*)ws + WS_WT);
  const float* ydot1 = (const float*)((const char*)ws + WS_YDOT1);
  const float* c1    = (const float*)((const char*)ws + WS_C1);

  const int wave  = threadIdx.x >> 5;
  const int lane  = threadIdx.x & 31;
  const int half  = lane >> 4;          // K-half select (16-bit A layout)
  const int mlane = lane & 15;          // A: row M ; B: col N ; C/D: col N
  const long long rowBase = (long long)blockIdx.x * 128 + wave * 16;

  const float* xrow = x + (rowBase + mlane) * DIM_EMBED;

  v8f acc0 = {}; v8f acc1 = {};
  union BU { v16bf v; uint4 q[2]; };

#pragma unroll
  for (int s = 0; s < 8; ++s) {         // K = 256 in steps of 32
    // A fragment: element j=g*8+w  <->  K = s*32 + g*16 + half*8 + w
    v16bf a;
#pragma unroll
    for (int g = 0; g < 2; ++g) {
      const int col = s * 32 + g * 16 + half * 8;
      const float4* p = (const float4*)(xrow + col);
      float4 f0 = p[0], f1 = p[1];
      a[g * 8 + 0] = (__bf16)f0.x; a[g * 8 + 1] = (__bf16)f0.y;
      a[g * 8 + 2] = (__bf16)f0.z; a[g * 8 + 3] = (__bf16)f0.w;
      a[g * 8 + 4] = (__bf16)f1.x; a[g * 8 + 5] = (__bf16)f1.y;
      a[g * 8 + 6] = (__bf16)f1.z; a[g * 8 + 7] = (__bf16)f1.w;
    }
    BU b0, b1;
#pragma unroll
    for (int g = 0; g < 2; ++g) {       // B loaded column-major == rows of W^T, same K split
      const int col = s * 32 + g * 16 + half * 8;
      b0.q[g] = *(const uint4*)(wt + (mlane)      * DIM_EMBED + col);
      b1.q[g] = *(const uint4*)(wt + (16 + mlane) * DIM_EMBED + col);
    }
    if (s < 7) __builtin_prefetch(xrow + (s + 1) * 32 + half * 8, 0, 0);

    acc0 = __builtin_amdgcn_wmma_f32_16x16x32_bf16(false, a, false, b0.v,
                                                   (short)0, acc0, false, false);
    acc1 = __builtin_amdgcn_wmma_f32_16x16x32_bf16(false, a, false, b1.v,
                                                   (short)0, acc1, false, false);
  }

  // C/D layout: VGPR r, lanes[0:16): M=r, N=lane ; lanes[16:32): M=8+r, N=lane-16
#pragma unroll
  for (int r = 0; r < 8; ++r) {
    const int M = r + 8 * half;
    const long long row = rowBase + M;
    const int dag = (int)(row / OPS_PER_DAG);
    float v0 = acc0[r] + ydot1[dag * H1V + mlane]      + c1[mlane];
    float v1 = acc1[r] + ydot1[dag * H1V + 16 + mlane] + c1[16 + mlane];
    h1lds[wave * 16 + M][mlane]      = fmaxf(v0, 0.f);
    h1lds[wave * 16 + M][mlane + 16] = fmaxf(v1, 0.f);
  }
  __syncthreads();

  const int t = threadIdx.x;
  if (t < 128) {                        // layers 2/3 in fp32 VALU (tiny)
    const long long row = (long long)blockIdx.x * 128 + t;
    float h2[H2V];
#pragma unroll
    for (int j = 0; j < H2V; ++j) {
      float a = opB2[j];
      for (int h = 0; h < H1V; ++h) a += h1lds[t][h] * opW2[h * H2V + j];
      h2[j] = fmaxf(a, 0.f);
    }
    float logit = opB3[0];
#pragma unroll
    for (int j = 0; j < H2V; ++j) logit += h2[j] * opW3[j];
    logit -= (1.f - op_msk[row]) * 1000.f;
    out_logits[row] = logit;
  }
}

// ---------------- global softmax over N=102400 (deterministic) ------------
__global__ void k_rmax1(const float* __restrict__ v, void* ws) {
  __shared__ float sd[256];
  int t = threadIdx.x;
  float m = -3.0e38f;
  for (int i = blockIdx.x * 256 + t; i < TOTAL_OPS; i += 256 * 256) m = fmaxf(m, v[i]);
  sd[t] = m; __syncthreads();
  for (int off = 128; off > 0; off >>= 1) {
    if (t < off) sd[t] = fmaxf(sd[t], sd[t + off]);
    __syncthreads();
  }
  if (t == 0) ((float*)((char*)ws + WS_PMAX))[blockIdx.x] = sd[0];
}

__global__ void k_rmax2(void* ws) {
  __shared__ float sd[256];
  int t = threadIdx.x;
  sd[t] = ((const float*)((const char*)ws + WS_PMAX))[t]; __syncthreads();
  for (int off = 128; off > 0; off >>= 1) {
    if (t < off) sd[t] = fmaxf(sd[t], sd[t + off]);
    __syncthreads();
  }
  if (t == 0) *(float*)((char*)ws + WS_GMAX) = sd[0];
}

__global__ void k_exp(float* __restrict__ v, void* ws) {
  __shared__ float sd[256];
  const float gmax = *(const float*)((const char*)ws + WS_GMAX);
  int t = threadIdx.x;
  float s = 0.f;
  for (int i = blockIdx.x * 256 + t; i < TOTAL_OPS; i += 256 * 256) {
    float e = expf(v[i] - gmax);
    v[i] = e;
    s += e;
  }
  sd[t] = s; __syncthreads();
  for (int off = 128; off > 0; off >>= 1) {
    if (t < off) sd[t] += sd[t + off];
    __syncthreads();
  }
  if (t == 0) ((float*)((char*)ws + WS_PSUM))[blockIdx.x] = sd[0];
}

__global__ void k_rsum2(void* ws) {
  __shared__ float sd[256];
  int t = threadIdx.x;
  sd[t] = ((const float*)((const char*)ws + WS_PSUM))[t]; __syncthreads();
  for (int off = 128; off > 0; off >>= 1) {
    if (t < off) sd[t] += sd[t + off];
    __syncthreads();
  }
  if (t == 0) *(float*)((char*)ws + WS_GINV) = 1.f / sd[0];
}

__global__ void k_scale(float* __restrict__ v, const void* ws) {
  const float inv = *(const float*)((const char*)ws + WS_GINV);
  int i = blockIdx.x * 256 + threadIdx.x;   // grid covers N exactly (400*256)
  v[i] *= inv;
}

// ---------------- prlvl branch: per-(dag,worker) MLP + 64-wide softmax ----
__global__ void k_prlvl(const float* __restrict__ prW1, const float* __restrict__ prW2,
                        const float* __restrict__ prB2, const float* __restrict__ prW3,
                        const float* __restrict__ prB3, const float* __restrict__ msk,
                        const void* __restrict__ ws, float* __restrict__ out) {
  __shared__ float sred[NUM_WORKERS];
  const float* ydotp = (const float*)((const char*)ws + WS_YDOTP);
  const float* cp1   = (const float*)((const char*)ws + WS_CP1);
  const int d = blockIdx.x, w = threadIdx.x;
  const float limit = (float)(w + 1);

  float h1[H1V];
#pragma unroll
  for (int h = 0; h < H1V; ++h)         // row 0 of pr_W1 multiplies limit
    h1[h] = fmaxf(limit * prW1[h] + ydotp[d * H1V + h] + cp1[h], 0.f);

  float h2[H2V];
#pragma unroll
  for (int j = 0; j < H2V; ++j) {
    float a = prB2[j];
    for (int h = 0; h < H1V; ++h) a += h1[h] * prW2[j + h * H2V];
    h2[j] = fmaxf(a, 0.f);
  }
  float logit = prB3[0];
#pragma unroll
  for (int j = 0; j < H2V; ++j) logit += h2[j] * prW3[j];
  logit -= (1.f - msk[d * NUM_WORKERS + w]) * 1000.f;

  // softmax over the 64 workers (fixed-order tree reductions)
  sred[w] = logit; __syncthreads();
  for (int off = 32; off > 0; off >>= 1) {
    if (w < off) sred[w] = fmaxf(sred[w], sred[w + off]);
    __syncthreads();
  }
  const float m = sred[0]; __syncthreads();
  const float e = expf(logit - m);
  sred[w] = e; __syncthreads();
  for (int off = 32; off > 0; off >>= 1) {
    if (w < off) sred[w] += sred[w + off];
    __syncthreads();
  }
  out[TOTAL_OPS + d * NUM_WORKERS + w] = e / sred[0];
}

// ---------------- launcher -------------------------------------------------
extern "C" void kernel_launch(void* const* d_in, const int* in_sizes, int n_in,
                              void* d_out, int out_size, void* d_ws, size_t ws_size,
                              hipStream_t stream) {
  (void)in_sizes; (void)n_in; (void)out_size; (void)ws_size;
  // input order from setup_inputs():
  // 0 num_ops, 1 num_dags, 2 num_workers, 3 x, 4 y, 5 z, 6 op_msk, 7 prlvl_msk,
  // 8 op_W1, 9 op_b1, 10 op_W2, 11 op_b2, 12 op_W3, 13 op_b3,
  // 14 pr_W1, 15 pr_b1, 16 pr_W2, 17 pr_b2, 18 pr_W3, 19 pr_b3
  const float* x       = (const float*)d_in[3];
  const float* y       = (const float*)d_in[4];
  const float* z       = (const float*)d_in[5];
  const float* op_msk  = (const float*)d_in[6];
  const float* pr_msk  = (const float*)d_in[7];
  const float* opW1 = (const float*)d_in[8];
  const float* opB1 = (const float*)d_in[9];
  const float* opW2 = (const float*)d_in[10];
  const float* opB2 = (const float*)d_in[11];
  const float* opW3 = (const float*)d_in[12];
  const float* opB3 = (const float*)d_in[13];
  const float* prW1 = (const float*)d_in[14];
  const float* prB1 = (const float*)d_in[15];
  const float* prW2 = (const float*)d_in[16];
  const float* prB2 = (const float*)d_in[17];
  const float* prW3 = (const float*)d_in[18];
  const float* prB3 = (const float*)d_in[19];
  float* out = (float*)d_out;

  k_const <<<1, 256, 0, stream>>>(z, opW1, opB1, prW1, prB1, d_ws);
  k_ydot  <<<NUM_DAGS, 64, 0, stream>>>(y, opW1, prW1, d_ws);
  k_gemm  <<<TOTAL_OPS / 128, 256, 0, stream>>>(x, op_msk, opW2, opB2, opW3, opB3,
                                                d_ws, out);
  k_rmax1 <<<256, 256, 0, stream>>>(out, d_ws);
  k_rmax2 <<<1, 256, 0, stream>>>(d_ws);
  k_exp   <<<256, 256, 0, stream>>>(out, d_ws);
  k_rsum2 <<<1, 256, 0, stream>>>(d_ws);
  k_scale <<<TOTAL_OPS / 256, 256, 0, stream>>>(out, d_ws);
  k_prlvl <<<NUM_DAGS, NUM_WORKERS, 0, stream>>>(prW1, prW2, prB2, prW3, prB3,
                                                 pr_msk, d_ws, out);
}
